// AttentionLayer_43516608643145
// MI455X (gfx1250) — matrix operand order repaired and checked by reading, not compiled, not run on hardware
//
#include <hip/hip_runtime.h>
#include <hip/hip_bf16.h>
#include <math.h>

typedef __attribute__((ext_vector_type(16))) __bf16 v16bf;
typedef __attribute__((ext_vector_type(8)))  float  v8f;
typedef __attribute__((ext_vector_type(4)))  float  f32x4;

#define B_  32
#define TE_ 8192
#define H_  128

#define ROWS_PER_WAVE 16
#define WAVES 8
#define ROWS_PER_ITER (ROWS_PER_WAVE * WAVES)   // 128
#define ITERS 4
#define ROWS_PER_BLOCK (ROWS_PER_ITER * ITERS)  // 512
#define CTX_CHUNK 512

// Native CDNA5 v_tanh_f32 (TRANS op) instead of libm's branchy exp-based tanh.
static __device__ __forceinline__ float fast_tanh(float x) {
#if __has_builtin(__builtin_amdgcn_tanhf)
  return __builtin_amdgcn_tanhf(x);
#else
  float y;
  // v_nop covers the TRANS-op result-use hazard window.
  asm("v_tanh_f32 %0, %1\n\tv_nop" : "=v"(y) : "v"(x));
  return y;
#endif
}

// ---------------------------------------------------------------------------
// Kernel 1: Uh[b,k] = sum_h dec[b,h] * U[h,k]; also zero the c_i output region
// (c_i is accumulated with atomics by context_kernel each call).
// ---------------------------------------------------------------------------
__global__ __launch_bounds__(H_)
void uh_kernel(const float* __restrict__ dec, const float* __restrict__ U,
               float* __restrict__ uh, float* __restrict__ c_out) {
  const int b = blockIdx.x;
  const int k = threadIdx.x;
  const float* d = dec + b * H_;
  float acc = 0.f;
#pragma unroll 8
  for (int h = 0; h < H_; ++h) acc = fmaf(d[h], U[h * H_ + k], acc);
  uh[b * H_ + k]    = acc;
  c_out[b * H_ + k] = 0.f;
}

// ---------------------------------------------------------------------------
// Kernel 2: raw scores[b,e] = tanh(enc[b,e,:]@W + Uh[b,:]) . V
// bf16 WMMA (f32 accumulate), fused epilogue with native v_tanh_f32.
// Scores land in the e_i output region; softmax_kernel normalizes in place.
// ---------------------------------------------------------------------------
__global__ __launch_bounds__(256, 2)
void scores_kernel(const float* __restrict__ enc, const float* __restrict__ W,
                   const float* __restrict__ uh, const float* __restrict__ V,
                   float* __restrict__ scores) {
  // W_a staged as bf16 in B-fragment order: frag(kk,t): lane l holds N = t*16 + (l&15),
  // halves i -> K = kk*32 + (i&7) + 8*(l>>4) + 16*(i>>3). 32B contiguous per lane.
  __shared__ __align__(32) __bf16 Wfrag[4 * 8 * 32 * 16];  // 32 KB
  __shared__ float uh_s[H_];
  __shared__ float v_s[H_];

  const int tid = threadIdx.x;
  const int b = blockIdx.y;
  const int row_block = blockIdx.x * ROWS_PER_BLOCK;

  for (int idx = tid; idx < 128 * 128; idx += 256) {
    const int i  = idx & 15;
    const int l  = (idx >> 4) & 31;
    const int t  = (idx >> 9) & 7;
    const int kk = idx >> 12;
    Wfrag[idx] = (__bf16)W[(kk * 32 + (i & 7) + ((l >> 4) << 3) + ((i >> 3) << 4)) * H_ + t * 16 + (l & 15)];
  }
  if (tid < H_) { uh_s[tid] = uh[b * H_ + tid]; v_s[tid] = V[tid]; }
  __syncthreads();

  const int lane  = tid & 31;
  const int wave  = tid >> 5;
  const int half  = lane >> 4;   // 0: rows M=0..7 / K-halves low; 1: M=8..15
  const int lrow  = lane & 15;
  const int khalf = half * 8;

  const v16bf* wlds = (const v16bf*)Wfrag;

  for (int it = 0; it < ITERS; ++it) {
    const int tile_row = row_block + it * ROWS_PER_ITER + wave * ROWS_PER_WAVE;
    const float* rowptr = enc + ((size_t)b * TE_ + tile_row + lrow) * H_;

    if (it + 1 < ITERS)  // warm L2/L0 for the next iteration's rows
      __builtin_prefetch(rowptr + (size_t)ROWS_PER_ITER * H_, 0, 0);

    // A fragments (16x32 bf16 per K-step), built per the documented layout:
    // lane<16: halves 0..7 -> K=kk*32+0..7, halves 8..15 -> K=kk*32+16..23
    // lane>=16: same +8.
    v16bf a[4];
#pragma unroll
    for (int kk = 0; kk < 4; ++kk) {
      const float* p = rowptr + kk * 32 + khalf;
      const f32x4 f0 = *(const f32x4*)(p);
      const f32x4 f1 = *(const f32x4*)(p + 4);
      const f32x4 f2 = *(const f32x4*)(p + 16);
      const f32x4 f3 = *(const f32x4*)(p + 20);
#pragma unroll
      for (int j = 0; j < 4; ++j) {
        a[kk][j]      = (__bf16)f0[j];
        a[kk][4 + j]  = (__bf16)f1[j];
        a[kk][8 + j]  = (__bf16)f2[j];
        a[kk][12 + j] = (__bf16)f3[j];
      }
    }

    float partial[8] = {0.f, 0.f, 0.f, 0.f, 0.f, 0.f, 0.f, 0.f};

#pragma unroll
    for (int t = 0; t < 8; ++t) {
      v8f acc = {};
#pragma unroll
      for (int kk = 0; kk < 4; ++kk) {
        const v16bf bfr = wlds[(kk * 8 + t) * 32 + lane];
        acc = __builtin_amdgcn_wmma_f32_16x16x32_bf16(
            /*neg_a=*/false, a[kk], /*neg_b=*/false, bfr,
            /*c_mod=*/(short)0, acc, /*reuse_a=*/false, /*reuse_b=*/false);
      }
      // D layout: lane l, vgpr v -> (M = v + 8*half, N = t*16 + lrow)
      const float un = uh_s[t * 16 + lrow];
      const float vn = v_s[t * 16 + lrow];
#pragma unroll
      for (int v = 0; v < 8; ++v)
        partial[v] = fmaf(fast_tanh(acc[v] + un), vn, partial[v]);
    }

    // Reduce each row over its 16-lane half-group; store raw score.
#pragma unroll
    for (int v = 0; v < 8; ++v) {
      float s = partial[v];
      s += __shfl_xor(s, 1, 16);
      s += __shfl_xor(s, 2, 16);
      s += __shfl_xor(s, 4, 16);
      s += __shfl_xor(s, 8, 16);
      if (lrow == v)
        scores[(size_t)b * TE_ + tile_row + v + khalf] = s;
    }
  }
}

// ---------------------------------------------------------------------------
// Kernel 3: in-place softmax over TE per batch (scores held in registers).
// ---------------------------------------------------------------------------
__global__ __launch_bounds__(256)
void softmax_kernel(float* __restrict__ e) {
  __shared__ float red[256];
  const int b = blockIdx.x;
  const int tid = threadIdx.x;
  float* row = e + (size_t)b * TE_;

  float vals[TE_ / 256];
  float m = -INFINITY;
#pragma unroll
  for (int i = 0; i < TE_ / 256; ++i) {
    vals[i] = row[i * 256 + tid];
    m = fmaxf(m, vals[i]);
  }
  red[tid] = m; __syncthreads();
  for (int s2 = 128; s2 > 0; s2 >>= 1) {
    if (tid < s2) red[tid] = fmaxf(red[tid], red[tid + s2]);
    __syncthreads();
  }
  m = red[0]; __syncthreads();

  float sum = 0.f;
#pragma unroll
  for (int i = 0; i < TE_ / 256; ++i) {
    vals[i] = __expf(vals[i] - m);
    sum += vals[i];
  }
  red[tid] = sum; __syncthreads();
  for (int s2 = 128; s2 > 0; s2 >>= 1) {
    if (tid < s2) red[tid] += red[tid + s2];
    __syncthreads();
  }
  const float inv = 1.f / red[0];
#pragma unroll
  for (int i = 0; i < TE_ / 256; ++i)
    row[i * 256 + tid] = vals[i] * inv;
}

// ---------------------------------------------------------------------------
// Kernel 4: c[b,h] += sum_te e[b,te] * enc[b,te,h].  Second encoder pass —
// encoder (134 MB) fits the 192 MB L2, so this should mostly hit cache.
// ---------------------------------------------------------------------------
__global__ __launch_bounds__(256)
void context_kernel(const float* __restrict__ enc, const float* __restrict__ e,
                    float* __restrict__ c) {
  const int b = blockIdx.y;
  const int chunk = blockIdx.x * CTX_CHUNK;
  const int tid = threadIdx.x;
  const int h  = tid & (H_ - 1);
  const int r0 = tid >> 7;  // 0 or 1: two TE rows in flight per block
  const float* ew = e + (size_t)b * TE_ + chunk;
  const float* eb = enc + ((size_t)b * TE_ + chunk) * H_;

  float acc = 0.f;
  for (int te = r0; te < CTX_CHUNK; te += 2)
    acc = fmaf(ew[te], eb[(size_t)te * H_ + h], acc);

  __shared__ float red[256];
  red[tid] = acc; __syncthreads();
  if (tid < H_)
    unsafeAtomicAdd(&c[b * H_ + h], red[tid] + red[tid + H_]);
}

// ---------------------------------------------------------------------------
extern "C" void kernel_launch(void* const* d_in, const int* in_sizes, int n_in,
                              void* d_out, int out_size, void* d_ws, size_t ws_size,
                              hipStream_t stream) {
  const float* enc = (const float*)d_in[0];  // [B, TE, H]
  const float* dec = (const float*)d_in[1];  // [B, 1, H]
  const float* W   = (const float*)d_in[2];  // [H, H]
  const float* U   = (const float*)d_in[3];  // [H, H]
  const float* V   = (const float*)d_in[4];  // [H, 1]

  float* e_out = (float*)d_out;              // [B, TE]
  float* c_out = e_out + (size_t)B_ * TE_;   // [B, H]
  float* uh    = (float*)d_ws;               // [B, H] scratch

  uh_kernel<<<dim3(B_), dim3(H_), 0, stream>>>(dec, U, uh, c_out);
  scores_kernel<<<dim3(TE_ / ROWS_PER_BLOCK, B_), dim3(256), 0, stream>>>(
      enc, W, uh, V, e_out);
  softmax_kernel<<<dim3(B_), dim3(256), 0, stream>>>(e_out);
  context_kernel<<<dim3(TE_ / CTX_CHUNK, B_), dim3(256), 0, stream>>>(
      enc, e_out, c_out);
}
